// MultiRegionRNN_59785944761036
// MI455X (gfx1250) — compile-verified, matrix operand order checked by reading, not run on hardware
//
#include <hip/hip_runtime.h>
#include <hip/hip_bf16.h>
#include <math.h>

typedef __attribute__((ext_vector_type(16))) _Float16 v16h;
typedef __attribute__((ext_vector_type(8)))  float    v8f;

#define NU   512        // units per region
#define BB   128        // batch
#define TT   500        // timesteps
#define DIN  10
#define DOUT 2
#define KCAT 1024       // concatenated K (r_m1 ; r_pmd)
#define NKT  32         // KCAT / 32
#define NNT  32         // NU / 16 n-tiles
#define NMT  8          // BB / 16 m-tiles
#define ABUF_HALVES (NMT * NKT * 32 * 16)   // 131072 = 128*1024
#define ASTRIP_HALVES (NKT * 32 * 16)       // 16384 halves = 32 KB per mtile

// ---------------------------------------------------------------------------
// Fragment index helpers (ISA 7.12.2, wave32):
//   A (16x32 f16, MxK): lane<16 -> M=lane, K = (h<8?0:16)+(h%8)
//                       lane>=16 -> M=lane-16, K = (h<8?0:16)+8+(h%8)
//   B (32x16 f16, KxN): same K mapping, N = lane%16
//   C/D (16x16 f32):    M = f + 8*(lane>>4), N = lane%16
// ---------------------------------------------------------------------------

// Pack concatenated weights into B-fragment layout:
//   wbuf[region][kt][ntile][lane][h], value = Wcat[region][k][n]
//   region 0 (M1):  k<512 -> W_rec_M1[n,k] + W_fb[n,:]@W_out[:,k] ; else W_pmd_m1[n,k-512]
//   region 1 (PMd): k<512 -> W_m1_pmd[n,k]                        ; else W_rec_PMd[n,k-512]
__global__ void pack_weights_kernel(const float* __restrict__ Wrec_m1,
                                    const float* __restrict__ Wrec_pmd,
                                    const float* __restrict__ Wpmd_m1,
                                    const float* __restrict__ Wm1_pmd,
                                    const float* __restrict__ Wout,
                                    const float* __restrict__ Wfb,
                                    _Float16* __restrict__ wbuf) {
    int idx = blockIdx.x * blockDim.x + threadIdx.x;   // 0 .. 2*1024*512-1
    if (idx >= 2 * KCAT * NU) return;
    int n      = idx & (NU - 1);
    int k      = (idx >> 9) & (KCAT - 1);
    int region = idx >> 19;
    float v;
    if (region == 0) {
        if (k < NU)
            v = Wrec_m1[n * NU + k] + Wfb[n * 2 + 0] * Wout[k] + Wfb[n * 2 + 1] * Wout[NU + k];
        else
            v = Wpmd_m1[n * NU + (k - NU)];
    } else {
        if (k < NU)
            v = Wm1_pmd[n * NU + k];
        else
            v = Wrec_pmd[n * NU + (k - NU)];
    }
    int kt   = k >> 5;
    int kl   = k & 31;
    int lane = (n & 15) + (((kl >> 3) & 1) ? 16 : 0);
    int h    = (kl & 7) + (((kl >> 4) & 1) ? 8 : 0);
    size_t off = ((((size_t)(region * NKT + kt)) * NNT + (n >> 4)) * 32 + lane) * 16 + h;
    wbuf[off] = (_Float16)v;
}

// Zero initial state: abuf ping-0 (rates at t=0 == 0), x state, and t=0 rates outputs.
__global__ void init_kernel(_Float16* __restrict__ abuf0,
                            float* __restrict__ xall,        // x_m1 then x_pmd, 2*128*512
                            float* __restrict__ rm1_0,
                            float* __restrict__ rpmd_0) {
    int i = blockIdx.x * blockDim.x + threadIdx.x;
    if (i < 2 * BB * NU) xall[i] = 0.0f;
    if (i < ABUF_HALVES) abuf0[i] = (_Float16)0.0f;
    if (i < BB * NU) { rm1_0[i] = 0.0f; rpmd_0[i] = 0.0f; }
}

// One recurrent step, both regions. Grid = 64 blocks x 256 thr (8 waves).
// Block id decodes (region, mtile, ntile-group) from SGPRs so the region
// branch is scalar. All 8 waves share one mtile: the 16x1024 A strip (32 KB)
// is staged in LDS once and re-read via ds_load; B streams from L2.
__global__ void __launch_bounds__(256)
rnn_step_kernel(const _Float16* __restrict__ abuf_prev,
                _Float16* __restrict__ abuf_next,
                const _Float16* __restrict__ wbuf,
                float* __restrict__ xall,                  // [2][128][512]
                const float* __restrict__ b_M1,
                const float* __restrict__ b_PMd,
                const float* __restrict__ u_prev,          // [128,10] = stim[t-1]
                const float* __restrict__ W_in_pmd,        // [512,10]
                float* __restrict__ rm1_t,                 // rates_m1 + t*B*N
                float* __restrict__ rpmd_t) {
    __shared__ _Float16 lAs[ASTRIP_HALVES];                // [kt][lane][h], 32 KB

    const int lane   = threadIdx.x & 31;
    const int wave   = threadIdx.x >> 5;
    const int region = blockIdx.x >> 5;                    // scalar: 0..1
    const int mtile  = (blockIdx.x >> 2) & 7;              // scalar: 0..7
    const int ntile  = (blockIdx.x & 3) * 8 + wave;        // 0..31

    // ---- cooperative load of this mtile's A strip into LDS (2048 x b128) ----
    {
        const uint4* src = (const uint4*)(abuf_prev + (size_t)mtile * ASTRIP_HALVES);
        uint4* dst = (uint4*)lAs;
#pragma unroll
        for (int c = 0; c < 8; ++c)
            dst[threadIdx.x + c * 256] = src[threadIdx.x + c * 256];
    }
    __syncthreads();

    // ---- GEMM: acc[16x16] = A(rates_prev)[mtile] x B(Wcat)[region][ntile] ----
    const v16h* aL = (const v16h*)lAs + lane;              // element stride = 32 lanes
    const _Float16* bp = wbuf + ((((size_t)region * NKT) * NNT + ntile) * 32 + lane) * 16;
    v8f acc = {};
#pragma unroll 4
    for (int kt = 0; kt < NKT; ++kt) {
        v16h a = aL[kt * 32];
        v16h b = *(const v16h*)bp;
        acc = __builtin_amdgcn_wmma_f32_16x16x32_f16(
            false, a, false, b, (short)0, acc, false, false);
        bp += NNT * 32 * 16;    // next kt within this region
    }

    // ---- epilogue: leaky integration + tanh, scatter to state/outputs ----
    const int nloc = lane & 15;
    const int nn   = ntile * 16 + nloc;
    const int mbase = (lane >> 4) * 8;
    const float bias = region ? b_PMd[nn] : b_M1[nn];

    // PMd-only input drive, guarded by a *scalar* branch (region from blockIdx)
    float uadd[8];
#pragma unroll
    for (int f = 0; f < 8; ++f) uadd[f] = 0.0f;
    if (region == 1) {
        float wj[DIN];
#pragma unroll
        for (int j = 0; j < DIN; ++j) wj[j] = W_in_pmd[nn * DIN + j];
#pragma unroll
        for (int f = 0; f < 8; ++f) {
            const int bb = mtile * 16 + f + mbase;
            float s = 0.0f;
#pragma unroll
            for (int j = 0; j < DIN; ++j) s += u_prev[bb * DIN + j] * wj[j];
            uadd[f] = s;
        }
    }

    float* xbuf = xall + (size_t)region * BB * NU;
    float* rout = region ? rpmd_t : rm1_t;

    // abuf_next scatter position: k index for next step = region*512 + nn
    const int kg  = region * NU + nn;
    const int kt2 = kg >> 5;
    const int kl  = kg & 31;
    const int lgrp = ((kl >> 3) & 1) ? 16 : 0;
    const int h2   = (kl & 7) + (((kl >> 4) & 1) ? 8 : 0);

#pragma unroll
    for (int f = 0; f < 8; ++f) {
        const int bb = mtile * 16 + f + mbase;
        const size_t xi = (size_t)bb * NU + nn;
        const float xn = 0.9f * xbuf[xi] + 0.1f * (acc[f] + bias + uadd[f]);
        xbuf[xi] = xn;
        const float r = tanhf(xn);
        rout[xi] = r;
        abuf_next[((((size_t)(bb >> 4) * NKT) + kt2) * 32 + ((bb & 15) + lgrp)) * 16 + h2] =
            (_Float16)r;
    }
}

// outs[t,b,:] = rates_m1[t,b,:] @ W_out.T  (batched over all T*B rows at the end)
__global__ void outs_kernel(const float* __restrict__ rates_m1,
                            const float* __restrict__ W_out,
                            float* __restrict__ outs, int nrows) {
    int row = blockIdx.x * blockDim.x + threadIdx.x;
    if (row >= nrows) return;
    const float* r = rates_m1 + (size_t)row * NU;
    float o0 = 0.0f, o1 = 0.0f;
    for (int k = 0; k < NU; ++k) {
        float rv = r[k];
        o0 += rv * W_out[k];
        o1 += rv * W_out[NU + k];
    }
    outs[row * 2 + 0] = o0;
    outs[row * 2 + 1] = o1;
}

extern "C" void kernel_launch(void* const* d_in, const int* in_sizes, int n_in,
                              void* d_out, int out_size, void* d_ws, size_t ws_size,
                              hipStream_t stream) {
    const float* stim     = (const float*)d_in[0];  // [500,128,10]
    const float* Wrec_m1  = (const float*)d_in[1];  // [512,512]
    const float* b_M1     = (const float*)d_in[2];  // [512]
    const float* Wrec_pmd = (const float*)d_in[3];  // [512,512]
    const float* b_PMd    = (const float*)d_in[4];  // [512]
    const float* Wpmd_m1  = (const float*)d_in[5];  // [512,512]
    const float* Wm1_pmd  = (const float*)d_in[6];  // [512,512]
    const float* Win_pmd  = (const float*)d_in[7];  // [512,10]
    const float* Wout     = (const float*)d_in[8];  // [2,512]
    const float* Wfb      = (const float*)d_in[9];  // [512,2]

    float* outs      = (float*)d_out;                          // [500,128,2]
    float* rates_m1  = outs + (size_t)TT * BB * DOUT;          // [500,128,512]
    float* rates_pmd = rates_m1 + (size_t)TT * BB * NU;        // [500,128,512]

    char* ws = (char*)d_ws;
    _Float16* wbuf = (_Float16*)ws;                                    // 2 MB
    _Float16* abuf = (_Float16*)(ws + (size_t)2 * KCAT * NU * 2);      // 2 x 256 KB ping-pong
    float*    xall = (float*)(ws + (size_t)2 * KCAT * NU * 2
                                 + (size_t)2 * ABUF_HALVES * 2);       // 512 KB

    // 1) fold feedback into W_rec_M1 and pack weights to B-fragment layout
    {
        int total = 2 * KCAT * NU;
        pack_weights_kernel<<<(total + 255) / 256, 256, 0, stream>>>(
            Wrec_m1, Wrec_pmd, Wpmd_m1, Wm1_pmd, Wout, Wfb, wbuf);
    }
    // 2) zero state, t=0 rates, and A-operand ping buffer 0
    {
        int total = 2 * BB * NU;   // 131072 covers all three zeroed regions
        init_kernel<<<(total + 255) / 256, 256, 0, stream>>>(
            abuf, xall, rates_m1, rates_pmd);
    }
    // 3) serial recurrence: one kernel per step, stream order = dependency
    for (int t = 1; t < TT; ++t) {
        const _Float16* aprev = abuf + (size_t)((t - 1) & 1) * ABUF_HALVES;
        _Float16*       anext = abuf + (size_t)(t & 1) * ABUF_HALVES;
        rnn_step_kernel<<<64, 256, 0, stream>>>(
            aprev, anext, wbuf, xall, b_M1, b_PMd,
            stim + (size_t)(t - 1) * BB * DIN, Win_pmd,
            rates_m1 + (size_t)t * BB * NU,
            rates_pmd + (size_t)t * BB * NU);
    }
    // 4) outputs for all timesteps from stored rates_m1 (t=0 rates are zero -> outs[0]=0)
    {
        int nrows = TT * BB;
        outs_kernel<<<(nrows + 255) / 256, 256, 0, stream>>>(rates_m1, Wout, outs, nrows);
    }
}
